// GeometricPenalty_23630910062961
// MI455X (gfx1250) — 8-bit matrix operand order repaired and checked by reading, compiled, not run on hardware
//
#include <hip/hip_runtime.h>


#define QLEVELS 32
#define ALPHA_C 0.1f
#define BETA_C 0.1f
#define MARGIN_C 10.0f
#define STEEP_C 10.0f
#define EPS_C 1e-8f

typedef int   v16i __attribute__((ext_vector_type(16)));
typedef float v8f  __attribute__((ext_vector_type(8)));
typedef unsigned int v4u __attribute__((ext_vector_type(4)));
typedef unsigned long long u64;
typedef unsigned char u8;

#if defined(__HIP_DEVICE_COMPILE__) && __has_builtin(__builtin_amdgcn_global_load_async_to_lds_b128)
#define GP_HAVE_ASYNC_LDS 1
#else
#define GP_HAVE_ASYNC_LDS 0
#endif

#if GP_HAVE_ASYNC_LDS
typedef int gp_gv4i __attribute__((vector_size(16)));
typedef __attribute__((address_space(1))) gp_gv4i* gp_glb_p;
typedef __attribute__((address_space(3))) gp_gv4i* gp_lds_p;
#define GP_GLB(p) ((gp_glb_p)(unsigned long long)(p))
#define GP_LDS(p) ((gp_lds_p)(unsigned int)(unsigned long long)(p))
__device__ __forceinline__ void gp_async_wait0() {
#if __has_builtin(__builtin_amdgcn_s_wait_asynccnt)
    __builtin_amdgcn_s_wait_asynccnt(0);
#else
    asm volatile("s_wait_asynccnt 0x0" ::: "memory");
#endif
}
#endif

__global__ void gp_zero(float* __restrict__ accum) {
    if (threadIdx.x < 4) accum[threadIdx.x] = 0.0f;
}

__global__ void gp_encode(const float* __restrict__ x, u8* __restrict__ codes,
                          float* __restrict__ rowQ, int D, int K) {
    __shared__ int sred[512];
    const int row = blockIdx.x;
    const int d   = threadIdx.x;
    float s = 1.0f / (1.0f + __expf(-STEEP_C * x[(size_t)row * D + d]));
    int q = (int)(s * (float)QLEVELS + 0.5f);
    q = q < 0 ? 0 : (q > QLEVELS ? QLEVELS : q);

    const u64 full = 0x3838383838383838ull;
    union { u64 q64[4]; v4u v[2]; } u;
#pragma unroll
    for (int c = 0; c < 4; ++c) {
        int qc = q - 8 * c;
        qc = qc < 0 ? 0 : (qc > 8 ? 8 : qc);
        u.q64[c] = qc ? (full >> ((8 - qc) * 8)) : 0ull;
    }
    u8* p = codes + (size_t)row * K + (size_t)d * QLEVELS;
    *(v4u*)(p)      = u.v[0];
    *(v4u*)(p + 16) = u.v[1];

    sred[d] = q;
    __syncthreads();
#pragma unroll
    for (int t = 256; t > 0; t >>= 1) {
        if (d < t && d + t < D) sred[d] += sred[d + t];
        __syncthreads();
    }
    if (d == 0) rowQ[row] = (float)sred[0];
}

#define BROW_V4 9
#define BSTRIP_V4 (64 * BROW_V4)

__global__ void __launch_bounds__(256)
gp_pairwise(const u8* __restrict__ codes, const float* __restrict__ rowQ,
            const int* __restrict__ labels, float* __restrict__ accum,
            int n, int K) {
    __shared__ v4u sB[2][BSTRIP_V4];

    const int tid   = threadIdx.x;
    const int lane  = tid & 31;
    const int wv    = tid >> 5;
    const int nBlks = n >> 6;
    const int bM    = blockIdx.x / nBlks;
    const int bN    = blockIdx.x - bM * nBlks;
    const int mBase = bM * 128 + wv * 16;
    const int nBase = bN * 64;
    const int hi    = (lane >> 4) & 1;
    const int lcol  = lane & 15;

    const int srow = tid >> 2;
    const int sseg = tid & 3;
    const u8* gsrc = codes + (size_t)(nBase + srow) * K + sseg * 32;
    v4u* sdst0 = &sB[0][srow * BROW_V4 + sseg * 2];
    v4u* sdst1 = &sB[1][srow * BROW_V4 + sseg * 2];

    const u8* aRow = codes + (size_t)(mBase + lcol) * K + (hi ? 16 : 0);

#if GP_HAVE_ASYNC_LDS
    __builtin_amdgcn_global_load_async_to_lds_b128(GP_GLB(gsrc),      GP_LDS(sdst0),     0, 0);
    __builtin_amdgcn_global_load_async_to_lds_b128(GP_GLB(gsrc + 16), GP_LDS(sdst0 + 1), 0, 0);
    gp_async_wait0();
#else
    sdst0[0] = *(const v4u*)(gsrc);
    sdst0[1] = *(const v4u*)(gsrc + 16);
#endif
    __syncthreads();

    v8f acc[4];
#pragma unroll
    for (int jt = 0; jt < 4; ++jt) acc[jt] = 0.0f;

    int p = 0;
    for (int k0 = 0; k0 < K; k0 += 128) {
        if (k0 + 128 < K) {
            const u8* gnx = gsrc + k0 + 128;
            v4u* dnx = p ? sdst0 : sdst1;
#if GP_HAVE_ASYNC_LDS
            __builtin_amdgcn_global_load_async_to_lds_b128(GP_GLB(gnx),      GP_LDS(dnx),     0, 0);
            __builtin_amdgcn_global_load_async_to_lds_b128(GP_GLB(gnx + 16), GP_LDS(dnx + 1), 0, 0);
#else
            dnx[0] = *(const v4u*)(gnx);
            dnx[1] = *(const v4u*)(gnx + 16);
#endif
        }

        union { v16i v; u64 q[8]; } a;
#pragma unroll
        for (int t = 0; t < 8; ++t) a.q[t] = *(const u64*)(aRow + k0 + 32 * (t >> 1) + 8 * (t & 1));

        const v4u* sp = sB[p];
#pragma unroll
        for (int jt = 0; jt < 4; ++jt) {
            union { v16i v; v4u q[4]; } b;
            const int rbase = (jt * 16 + lcol) * BROW_V4 + hi;
#pragma unroll
            for (int t = 0; t < 4; ++t) b.q[t] = sp[rbase + 2 * t];
            acc[jt] = __builtin_amdgcn_wmma_f32_16x16x128_fp8_fp8(
                a.v, b.v, (short)0, acc[jt], false, false);
        }

#if GP_HAVE_ASYNC_LDS
        gp_async_wait0();
#endif
        __syncthreads();
        p ^= 1;
    }

    float Qi[8];
    int   labi[8];
#pragma unroll
    for (int v = 0; v < 8; ++v) {
        int i = mBase + hi * 8 + v;
        Qi[v]   = rowQ[i];
        labi[v] = labels[i];
    }

    const float invQ = 1.0f / (float)QLEVELS;
    float intra_s = 0.0f, intra_c = 0.0f, inter_s = 0.0f, inter_c = 0.0f;
#pragma unroll
    for (int jt = 0; jt < 4; ++jt) {
        int j = nBase + jt * 16 + lcol;
        float Qj = rowQ[j];
        int labj = labels[j];
#pragma unroll
        for (int v = 0; v < 8; ++v) {
            int i = mBase + hi * 8 + v;
            float dist = (Qi[v] + Qj - 2.0f * acc[jt][v]) * invQ;
            if (i != j) {
                if (labi[v] == labj) {
                    intra_s += dist;
                    intra_c += 1.0f;
                } else {
                    float r = MARGIN_C - dist;
                    inter_s += (r > 0.0f) ? r : 0.0f;
                    inter_c += 1.0f;
                }
            }
        }
    }

#pragma unroll
    for (int off = 16; off > 0; off >>= 1) {
        intra_s += __shfl_xor(intra_s, off, 32);
        intra_c += __shfl_xor(intra_c, off, 32);
        inter_s += __shfl_xor(inter_s, off, 32);
        inter_c += __shfl_xor(inter_c, off, 32);
    }
    if (lane == 0) {
        atomicAdd(&accum[0], intra_s);
        atomicAdd(&accum[1], intra_c);
        atomicAdd(&accum[2], inter_s);
        atomicAdd(&accum[3], inter_c);
    }
}

__global__ void gp_final(const float* __restrict__ accum, float* __restrict__ out) {
    out[0] = ALPHA_C * accum[0] / (accum[1] + EPS_C) +
             BETA_C  * accum[2] / (accum[3] + EPS_C);
}

extern "C" void kernel_launch(void* const* d_in, const int* in_sizes, int n_in,
                              void* d_out, int out_size, void* d_ws, size_t ws_size,
                              hipStream_t stream) {
    const float* x      = (const float*)d_in[0];
    const int*   labels = (const int*)d_in[1];
    float*       out    = (float*)d_out;

    const int n = in_sizes[1];
    const int D = in_sizes[0] / n;
    const int K = D * QLEVELS;

    char*  ws    = (char*)d_ws;
    float* accum = (float*)ws;
    float* rowQ  = (float*)(ws + 256);
    size_t codesOff = 256 + (((size_t)n * 4 + 255) & ~(size_t)255);
    u8*    codes = (u8*)(ws + codesOff);

    gp_zero<<<1, 32, 0, stream>>>(accum);
    gp_encode<<<n, D, 0, stream>>>(x, codes, rowQ, D, K);

    const int blocks = (n / 128) * (n / 64);
    gp_pairwise<<<blocks, 256, 0, stream>>>(codes, rowQ, labels, accum, n, K);

    gp_final<<<1, 1, 0, stream>>>(accum, out);
}
